// AdaptiveMemoryCompressor_19490561589548
// MI455X (gfx1250) — matrix operand order repaired and checked by reading, not compiled, not run on hardware
//
#include <hip/hip_runtime.h>

#define B_ 8
#define L_ 4096
#define D_ 1024
#define H_ 16
#define HD_ 64
#define T_ 64
#define DFF_ 2048

typedef __attribute__((ext_vector_type(16))) __bf16 bf16x16;
typedef __attribute__((ext_vector_type(8)))  float  f32x8;
typedef __attribute__((ext_vector_type(4)))  unsigned uint32x4;
typedef __attribute__((ext_vector_type(8)))  int      int32x8;
typedef __attribute__((ext_vector_type(4)))  int      int32x4;

#if defined(__gfx1250__) && __has_builtin(__builtin_amdgcn_tensor_load_to_lds)
#define USE_TDM 1
#else
#define USE_TDM 0
#endif

union Frag { bf16x16 v; unsigned u[8]; uint4 q[2]; };

__device__ __forceinline__ f32x8 zero8() {
  f32x8 z = {0.f,0.f,0.f,0.f,0.f,0.f,0.f,0.f};
  return z;
}

__device__ __forceinline__ f32x8 wmma_bf16(const Frag& a, const Frag& b, f32x8 c) {
  return __builtin_amdgcn_wmma_f32_16x16x32_bf16(false, a.v, false, b.v, (short)0, c, false, false);
}

#if USE_TDM
// TDM: DMA one [rowsTile x 16-dword] tile (rows of 32 bf16) from a row-major
// tensor with row stride Kdw dwords into LDS laid out with 20-dword row stride
// (pad_interval = 16 dwords -> pad_amount = 4 dwords). Rows >= Mrows zero-fill.
__device__ __forceinline__ void tdm_load_tile(const void* gaddr, unsigned lds_byte,
                                              int Kdw, int Mrows, int rowsTile) {
  unsigned long long ga = (unsigned long long)(uintptr_t)gaddr;
  uint32x4 g0;
  g0[0] = 1u;                                   // count=1, user descriptor
  g0[1] = lds_byte;                             // lds_addr
  g0[2] = (unsigned)ga;                         // global_addr[31:0] (desc bits 95:64)
  g0[3] = (unsigned)(ga >> 32) | (2u << 30);    // global_addr[56:32] | type=2
  int32x8 g1;
  g1[0] = (int)((2u << 16) |                    // data_size = 4 bytes
                (1u << 20) |                    // pad_enable
                (3u << 22) |                    // pad_interval = 16 dwords
                (3u << 25));                    // pad_amount   = 4 dwords
  g1[1] = (int)(((unsigned)Kdw & 0xffffu) << 16);              // tensor_dim0[15:0]
  g1[2] = (int)(((unsigned)Kdw >> 16) |
                (((unsigned)Mrows & 0xffffu) << 16));          // dim0 hi | dim1 lo
  g1[3] = (int)(((unsigned)Mrows >> 16) | (16u << 16));        // dim1 hi | tile_dim0=16
  g1[4] = rowsTile;                                            // tile_dim1 (tile_dim2=0)
  g1[5] = Kdw;                                                 // tensor_dim0_stride lo
  g1[6] = 0;
  g1[7] = 0;
  int32x4 z4 = {0, 0, 0, 0};
  int32x8 z8 = {0, 0, 0, 0, 0, 0, 0, 0};
  __builtin_amdgcn_tensor_load_to_lds(g0, g1, z4, z4, z8, 0);
}
#endif

// ---------------------------------------------------------------- convert
__global__ void cvt_kernel(const float* __restrict__ in, __bf16* __restrict__ out, int n) {
  for (int i = blockIdx.x * blockDim.x + threadIdx.x; i < n; i += gridDim.x * blockDim.x)
    out[i] = (__bf16)in[i];
}

// ---------------------------------------------------------------- generic GEMM
// C[M,N] = X[M,K] @ W[N,K]^T + bias ; epi: 0 = f32 store, 1 = bf16 store, 2 = GELU->bf16
__global__ __launch_bounds__(256) void gemm_bf16_kernel(
    const __bf16* __restrict__ X, const __bf16* __restrict__ W,
    const float* __restrict__ bias,
    float* __restrict__ Cf, __bf16* __restrict__ Cb,
    int M, int N, int K, int epi)
{
  __shared__ unsigned sX[2][128][20];
  __shared__ unsigned sW[2][128][20];
  const int tid  = threadIdx.x;
  const int lane = tid & 31, wid = tid >> 5;
  const int half = lane >> 4, l16 = lane & 15;
  const int wm = wid >> 2, wn = wid & 3;            // 2 x 4 wave grid, each wave 64x32
  const int m0 = blockIdx.y * 128, n0 = blockIdx.x * 128;
  const int K8 = K >> 3;                            // uint4 per row

  const uint4* X4 = (const uint4*)X;
  const uint4* W4 = (const uint4*)W;
  (void)X4; (void)W4; (void)K8;

  f32x8 acc[4][2];
#pragma unroll
  for (int i = 0; i < 4; ++i)
#pragma unroll
    for (int j = 0; j < 2; ++j) acc[i][j] = zero8();

  auto stage = [&](int kt, int buf) {
#if USE_TDM
    // One wave DMAs the X tile, another the W tile; others just compute.
    if (wid == 0) {
      tdm_load_tile((const char*)X + ((size_t)m0 * K + (size_t)kt * 32) * 2,
                    (unsigned)(uintptr_t)&sX[buf][0][0], K >> 1, M - m0, 128);
    } else if (wid == 1) {
      tdm_load_tile((const char*)W + ((size_t)n0 * K + (size_t)kt * 32) * 2,
                    (unsigned)(uintptr_t)&sW[buf][0][0], K >> 1, 128, 128);
    }
#else
#pragma unroll
    for (int s = 0; s < 2; ++s) {
      int i = tid * 2 + s;            // 0..511
      int row = i >> 2, seg = i & 3;  // 128 rows x 4 uint4
      int xr = m0 + row; xr = xr < M ? xr : M - 1;
      uint4 dx = X4[(size_t)xr * K8 + kt * 4 + seg];
      *(uint4*)&sX[buf][row][seg * 4] = dx;
      int wr = n0 + row;              // N always multiple of 128 here
      uint4 dw = W4[(size_t)wr * K8 + kt * 4 + seg];
      *(uint4*)&sW[buf][row][seg * 4] = dw;
    }
#endif
  };

  const int KT = K >> 5;
  stage(0, 0);
#if USE_TDM
  __builtin_amdgcn_s_wait_tensorcnt(0);
#endif
  __syncthreads();
  for (int kt = 0; kt < KT; ++kt) {
    int cur = kt & 1;
    if (kt + 1 < KT) stage(kt + 1, cur ^ 1);
    Frag a[4], b[2];
#pragma unroll
    for (int fa = 0; fa < 4; ++fa) {
      int row = wm * 64 + fa * 16 + l16;
      a[fa].q[0] = *(uint4*)&sX[cur][row][half * 4];
      a[fa].q[1] = *(uint4*)&sX[cur][row][8 + half * 4];
    }
#pragma unroll
    for (int fb = 0; fb < 2; ++fb) {
      int row = wn * 32 + fb * 16 + l16;
      b[fb].q[0] = *(uint4*)&sW[cur][row][half * 4];
      b[fb].q[1] = *(uint4*)&sW[cur][row][8 + half * 4];
    }
#pragma unroll
    for (int fa = 0; fa < 4; ++fa)
#pragma unroll
      for (int fb = 0; fb < 2; ++fb)
        acc[fa][fb] = wmma_bf16(a[fa], b[fb], acc[fa][fb]);
#if USE_TDM
    __builtin_amdgcn_s_wait_tensorcnt(0);   // no-op for non-issuing waves
#endif
    __syncthreads();
  }

#pragma unroll
  for (int fa = 0; fa < 4; ++fa)
#pragma unroll
    for (int fb = 0; fb < 2; ++fb)
#pragma unroll
      for (int r = 0; r < 8; ++r) {
        int m = m0 + wm * 64 + fa * 16 + r + 8 * half;
        int n = n0 + wn * 32 + fb * 16 + l16;
        if (m < M) {
          float y = acc[fa][fb][r] + bias[n];
          if (epi == 0) {
            Cf[(size_t)m * N + n] = y;
          } else {
            if (epi == 2) y = 0.5f * y * (1.0f + erff(y * 0.70710678118654752f));
            Cb[(size_t)m * N + n] = (__bf16)y;
          }
        }
      }
}

// ---------------------------------------------------------------- fused attention
// One block per (b,h). 4 waves; each wave owns a disjoint L-range with
// independent online softmax + O accumulator; merged via LDS at the end.
__global__ __launch_bounds__(128) void attn_kernel(
    const __bf16* __restrict__ q,   // [64, 1024]
    const __bf16* __restrict__ kv,  // [B*L, 2048]  (K | V per head)
    __bf16* __restrict__ o)         // [B*T, 1024]
{
  const int bh = blockIdx.x;
  const int b = bh >> 4, h = bh & 15;
  const int tid = threadIdx.x;
  const int wid = tid >> 5, lane = tid & 31;
  const int half = lane >> 4, l16 = lane & 15;

  __shared__ unsigned qs[64][36];        // 64x64 bf16 (32 dwords + pad)
  __shared__ unsigned vts[4][64][17];    // per-wave V^T chunk: [d][l-pair dwords]
  __shared__ unsigned ps[4][64][17];     // per-wave P chunk (bf16)
  __shared__ float ms_s[4][64];
  __shared__ float ls_s[4][64];
  __shared__ float os[64][65];

  const unsigned* q32  = (const unsigned*)q;
  const unsigned* kv32 = (const unsigned*)kv;

  for (int i = tid; i < 64 * 65; i += 128) (&os[0][0])[i] = 0.f;
#pragma unroll
  for (int j = 0; j < 16; ++j) {
    int idx = tid + j * 128;            // 2048 dwords
    int row = idx >> 5, dw = idx & 31;
    qs[row][dw] = q32[row * 512 + h * 32 + dw];
  }
  __syncthreads();

  f32x8 O[4][4];
  float mrun[4][8], lrun[4][8];
#pragma unroll
  for (int i = 0; i < 4; ++i) {
#pragma unroll
    for (int j = 0; j < 4; ++j) O[i][j] = zero8();
#pragma unroll
    for (int r = 0; r < 8; ++r) { mrun[i][r] = -3.0e38f; lrun[i][r] = 0.f; }
  }

  const int iters = (L_ / 4) / 32;
  for (int it = 0; it < iters; ++it) {
    const int base = wid * (L_ / 4) + it * 32;

    // ---- S = q @ K^T over 32 memory rows (K dim = 64, two WMMA k-steps) ----
    f32x8 s[4][2];
#pragma unroll
    for (int i = 0; i < 4; ++i)
#pragma unroll
      for (int j = 0; j < 2; ++j) s[i][j] = zero8();
#pragma unroll
    for (int kk = 0; kk < 2; ++kk) {
      Frag a[4], bk[2];
#pragma unroll
      for (int fa = 0; fa < 4; ++fa) {
        int row = fa * 16 + l16;
        a[fa].q[0] = *(const uint4*)&qs[row][kk * 16 + half * 4];
        a[fa].q[1] = *(const uint4*)&qs[row][kk * 16 + 8 + half * 4];
      }
#pragma unroll
      for (int fb = 0; fb < 2; ++fb) {
        size_t gl = (size_t)(b * L_ + base + fb * 16 + l16) * 1024;
        const uint4* p4 = (const uint4*)&kv32[gl + h * 32 + kk * 16 + half * 4];
        bk[fb].q[0] = p4[0];
        bk[fb].q[1] = p4[2];
      }
#pragma unroll
      for (int fa = 0; fa < 4; ++fa)
#pragma unroll
        for (int fb = 0; fb < 2; ++fb)
          s[fa][fb] = wmma_bf16(a[fa], bk[fb], s[fa][fb]);
    }

    // ---- stage V^T for this chunk into per-wave LDS ----
    {
      int p = l16, dhalf = half;  // lane -> row-pair p, d-half
      size_t g0 = (size_t)(b * L_ + base + 2 * p) * 1024 + 512 + h * 32 + dhalf * 16;
      size_t g1 = g0 + 1024;
      unsigned va[16], vb[16];
#pragma unroll
      for (int j = 0; j < 4; ++j) {
        *(uint4*)&va[j * 4] = *(const uint4*)&kv32[g0 + j * 4];
        *(uint4*)&vb[j * 4] = *(const uint4*)&kv32[g1 + j * 4];
      }
#pragma unroll
      for (int e = 0; e < 32; ++e) {
        unsigned lo = (va[e >> 1] >> ((e & 1) * 16)) & 0xffffu;
        unsigned hi = (vb[e >> 1] >> ((e & 1) * 16)) & 0xffffu;
        vts[wid][dhalf * 32 + e][p] = lo | (hi << 16);
      }
    }

    // ---- online softmax; write P (bf16) to LDS; rescale O ----
#pragma unroll
    for (int fm = 0; fm < 4; ++fm)
#pragma unroll
      for (int r = 0; r < 8; ++r) {
        float v0 = s[fm][0][r] * 0.125f;   // 1/sqrt(hd)
        float v1 = s[fm][1][r] * 0.125f;
        float mx = fmaxf(v0, v1);
#pragma unroll
        for (int off = 1; off < 16; off <<= 1) mx = fmaxf(mx, __shfl_xor(mx, off, 32));
        float mnew = fmaxf(mrun[fm][r], mx);
        float corr = __expf(mrun[fm][r] - mnew);
        mrun[fm][r] = mnew;
        float p0 = __expf(v0 - mnew);
        float p1 = __expf(v1 - mnew);
        float rs = p0 + p1;
#pragma unroll
        for (int off = 1; off < 16; off <<= 1) rs += __shfl_xor(rs, off, 32);
        lrun[fm][r] = lrun[fm][r] * corr + rs;
#pragma unroll
        for (int fn = 0; fn < 4; ++fn) O[fm][fn][r] *= corr;
        int t = fm * 16 + r + 8 * half;
        __bf16* prow = (__bf16*)&ps[wid][t][0];
        prow[l16]      = (__bf16)p0;
        prow[16 + l16] = (__bf16)p1;
      }

    asm volatile("s_wait_dscnt 0" ::: "memory");

    // ---- O += P @ V (K=32, one WMMA k-step, 4x4 tiles) ----
    Frag pa[4], vbf[4];
#pragma unroll
    for (int fm = 0; fm < 4; ++fm) {
      int t = fm * 16 + l16;
#pragma unroll
      for (int v = 0; v < 8; ++v)
        pa[fm].u[v] = ps[wid][t][(v >> 2) * 8 + half * 4 + (v & 3)];
    }
#pragma unroll
    for (int fn = 0; fn < 4; ++fn) {
      int dr = fn * 16 + l16;
#pragma unroll
      for (int v = 0; v < 8; ++v)
        vbf[fn].u[v] = vts[wid][dr][(v >> 2) * 8 + half * 4 + (v & 3)];
    }
#pragma unroll
    for (int fm = 0; fm < 4; ++fm)
#pragma unroll
      for (int fn = 0; fn < 4; ++fn)
        O[fm][fn] = wmma_bf16(pa[fm], vbf[fn], O[fm][fn]);
  }

  // ---- merge wave partials (log-sum-exp) ----
  if (l16 == 0) {
#pragma unroll
    for (int fm = 0; fm < 4; ++fm)
#pragma unroll
      for (int r = 0; r < 8; ++r) {
        int t = fm * 16 + r + 8 * half;
        ms_s[wid][t] = mrun[fm][r];
        ls_s[wid][t] = lrun[fm][r];
      }
  }
  __syncthreads();
#pragma unroll
  for (int fm = 0; fm < 4; ++fm)
#pragma unroll
    for (int r = 0; r < 8; ++r) {
      int t = fm * 16 + r + 8 * half;
      float M = ms_s[0][t];
#pragma unroll
      for (int w = 1; w < 4; ++w) M = fmaxf(M, ms_s[w][t]);
      float sc = __expf(mrun[fm][r] - M);
#pragma unroll
      for (int fn = 0; fn < 4; ++fn)
        atomicAdd(&os[t][fn * 16 + l16], O[fm][fn][r] * sc);
    }
  __syncthreads();
  for (int idx = tid; idx < 64 * 64; idx += 128) {
    int t = idx >> 6, d = idx & 63;
    float M = ms_s[0][t];
#pragma unroll
    for (int w = 1; w < 4; ++w) M = fmaxf(M, ms_s[w][t]);
    float den = 0.f;
#pragma unroll
    for (int w = 0; w < 4; ++w) den += ls_s[w][t] * __expf(ms_s[w][t] - M);
    float val = os[t][d] / den;
    o[(size_t)(b * T_ + t) * D_ + h * HD_ + d] = (__bf16)val;
  }
}

// ---------------------------------------------------------------- layernorm
__global__ __launch_bounds__(256) void ln_kernel(const float* __restrict__ x,
    const float* __restrict__ g, const float* __restrict__ bb, float* __restrict__ out) {
  __shared__ float s1[256], s2[256];
  int row = blockIdx.x, tid = threadIdx.x;
  float a = 0.f, c = 0.f;
  for (int j = tid; j < D_; j += 256) {
    float v = x[(size_t)row * D_ + j];
    a += v; c += v * v;
  }
  s1[tid] = a; s2[tid] = c;
  __syncthreads();
  for (int st = 128; st > 0; st >>= 1) {
    if (tid < st) { s1[tid] += s1[tid + st]; s2[tid] += s2[tid + st]; }
    __syncthreads();
  }
  float mu = s1[0] * (1.f / D_);
  float var = s2[0] * (1.f / D_) - mu * mu;
  float rstd = rsqrtf(var + 1e-5f);
  for (int j = tid; j < D_; j += 256)
    out[(size_t)row * D_ + j] = (x[(size_t)row * D_ + j] - mu) * rstd * g[j] + bb[j];
}

// ---------------------------------------------------------------- linear interp T -> L (+bf16)
__global__ __launch_bounds__(256) void interp_kernel(const float* __restrict__ comp,
                                                     __bf16* __restrict__ up) {
  for (int row = blockIdx.x; row < B_ * L_; row += gridDim.x) {
    int b = row / L_, l = row % L_;
    float pos = (l + 0.5f) * ((float)T_ / (float)L_) - 0.5f;
    pos = fminf(fmaxf(pos, 0.f), (float)(T_ - 1));
    int i0 = (int)floorf(pos);
    int i1 = (i0 + 1 < T_) ? i0 + 1 : T_ - 1;
    float f = pos - (float)i0;
    const float* c0 = comp + (size_t)(b * T_ + i0) * D_;
    const float* c1 = comp + (size_t)(b * T_ + i1) * D_;
    for (int d = threadIdx.x; d < D_; d += 256)
      up[(size_t)row * D_ + d] = (__bf16)(c0[d] * (1.f - f) + c1[d] * f);
  }
}

// ---------------------------------------------------------------- driver
extern "C" void kernel_launch(void* const* d_in, const int* in_sizes, int n_in,
                              void* d_out, int out_size, void* d_ws, size_t ws_size,
                              hipStream_t stream) {
  (void)in_sizes; (void)n_in; (void)out_size; (void)ws_size;
  const float* memory = (const float*)d_in[0];
  const float* query  = (const float*)d_in[1];
  const float* in_w   = (const float*)d_in[2];
  const float* in_b   = (const float*)d_in[3];
  const float* ow     = (const float*)d_in[4];
  const float* ob     = (const float*)d_in[5];
  const float* ln_g   = (const float*)d_in[6];
  const float* ln_b   = (const float*)d_in[7];
  const float* w1     = (const float*)d_in[8];
  const float* b1     = (const float*)d_in[9];
  const float* w2     = (const float*)d_in[10];
  const float* b2     = (const float*)d_in[11];

  float* compressed = (float*)d_out;                       // [B,T,D]
  float* recon = compressed + (size_t)B_ * T_ * D_;        // [B,L,D]

  char* ws = (char*)d_ws;
  size_t off = 0;
  auto alloc = [&](size_t bytes) {
    void* p = ws + off;
    off += (bytes + 255) & ~(size_t)255;
    return p;
  };
  __bf16* mem_b = (__bf16*)alloc((size_t)B_ * L_ * D_ * 2);      // later reused as `up`
  __bf16* kv_b  = (__bf16*)alloc((size_t)B_ * L_ * 2 * D_ * 2);  // later reused as `h`
  __bf16* qx_b  = (__bf16*)alloc((size_t)T_ * D_ * 2);
  __bf16* qp_b  = (__bf16*)alloc((size_t)T_ * D_ * 2);
  __bf16* inw_b = (__bf16*)alloc((size_t)3 * D_ * D_ * 2);
  __bf16* ow_b  = (__bf16*)alloc((size_t)D_ * D_ * 2);
  __bf16* w1_b  = (__bf16*)alloc((size_t)DFF_ * D_ * 2);
  __bf16* w2_b  = (__bf16*)alloc((size_t)D_ * DFF_ * 2);
  __bf16* at_b  = (__bf16*)alloc((size_t)B_ * T_ * D_ * 2);
  float*  pj_f  = (float*)alloc((size_t)B_ * T_ * D_ * 4);

  cvt_kernel<<<4096, 256, 0, stream>>>(memory, mem_b, B_ * L_ * D_);
  cvt_kernel<<<64,   256, 0, stream>>>(query,  qx_b,  T_ * D_);
  cvt_kernel<<<2048, 256, 0, stream>>>(in_w,   inw_b, 3 * D_ * D_);
  cvt_kernel<<<1024, 256, 0, stream>>>(ow,     ow_b,  D_ * D_);
  cvt_kernel<<<1024, 256, 0, stream>>>(w1,     w1_b,  DFF_ * D_);
  cvt_kernel<<<1024, 256, 0, stream>>>(w2,     w2_b,  D_ * DFF_);

  // q projection: [64,1024] = query @ Wq^T + bq  -> bf16
  gemm_bf16_kernel<<<dim3(D_ / 128, 1), 256, 0, stream>>>(
      qx_b, inw_b, in_b, nullptr, qp_b, T_, D_, D_, 1);
  // fused K|V projection: [32768,2048] = memory @ [Wk;Wv]^T + [bk;bv] -> bf16
  gemm_bf16_kernel<<<dim3(2 * D_ / 128, B_ * L_ / 128), 256, 0, stream>>>(
      mem_b, inw_b + (size_t)D_ * D_, in_b + D_, nullptr, kv_b, B_ * L_, 2 * D_, D_, 1);
  // fused flash attention -> bf16 [B*T, D]
  attn_kernel<<<B_ * H_, 128, 0, stream>>>(qp_b, kv_b, at_b);
  // out_proj -> f32
  gemm_bf16_kernel<<<dim3(D_ / 128, B_ * T_ / 128), 256, 0, stream>>>(
      at_b, ow_b, ob, pj_f, nullptr, B_ * T_, D_, D_, 0);
  // LayerNorm -> compressed (output 1)
  ln_kernel<<<B_ * T_, 256, 0, stream>>>(pj_f, ln_g, ln_b, compressed);
  // interpolate T->L, cast bf16 (reuse mem_b)
  interp_kernel<<<4096, 256, 0, stream>>>(compressed, mem_b);
  // MLP1: h = GELU(up @ w1^T + b1) -> bf16 (reuse kv_b)
  gemm_bf16_kernel<<<dim3(DFF_ / 128, B_ * L_ / 128), 256, 0, stream>>>(
      mem_b, w1_b, b1, nullptr, kv_b, B_ * L_, DFF_, D_, 2);
  // MLP2: recon = h @ w2^T + b2 -> f32 (output 2)
  gemm_bf16_kernel<<<dim3(D_ / 128, B_ * L_ / 128), 256, 0, stream>>>(
      kv_b, w2_b, b2, recon, nullptr, B_ * L_, D_, DFF_, 0);
}
